// Model_4681514352772
// MI455X (gfx1250) — matrix operand order repaired and checked by reading, not compiled, not run on hardware
//
#include <hip/hip_runtime.h>
#include <math.h>

#define D_MODEL 768
#define NHEAD   12
#define HDIM    64
#define FFN     3072
#define S_ENC   1024
#define T_DEC   512
#define VOCAB   50265
#define ATT_SCALE 0.125f   // HD^-0.5
#define NEGBIG  (-1.0e9f)

typedef __attribute__((ext_vector_type(16))) _Float16 v16h;
typedef __attribute__((ext_vector_type(8)))  _Float16 v8h;
typedef __attribute__((ext_vector_type(8)))  float    v8f;

// ---------------------------------------------------------------------------
// Tiled WMMA GEMM: C[M,N] = act(A[M,K] @ B + bias)
//   BT=false: B is [K,N] row-major (x @ w convention)
//   BT=true : B is [N,K] row-major (logits = y @ embed.T)
// Block = 256 threads (8 wave32), tile 64x64, K-step 32.
// Staging: two float4 global loads per thread per operand (b128, coalesced),
// register-prefetch of the next K-tile while WMMAs run, f16 LDS tiles with
// padded rows, fp32 accumulation in v_wmma_f32_16x16x32_f16.
// Grid mapping is (M/64, N/64) with x = row-block so consecutive blocks share
// the same B stripe through L2 (B-stationary for the HBM-bound logits GEMM).
// M must be a multiple of 64 (1024/512 here); K % 32 == 0; N arbitrary.
// ---------------------------------------------------------------------------
__device__ __forceinline__ float gelu_exact(float x) {
    return 0.5f * x * (1.0f + erff(x * 0.70710678118654752f));
}

template <int ACT, bool BT>
__global__ __launch_bounds__(256) void gemm_kernel(const float* __restrict__ A,
                                                   const float* __restrict__ B,
                                                   const float* __restrict__ bias,
                                                   float* __restrict__ C,
                                                   int M, int N, int K) {
    __shared__ __align__(16) _Float16 As[64][40];   // 64 m x 32 k (+8 pad)
    __shared__ __align__(16) _Float16 Bs[64][40];   // 64 n x 32 k (+8 pad), n-major

    const int MB   = blockIdx.x << 6;   // row block (x-fastest -> B reuse in L2)
    const int NB   = blockIdx.y << 6;
    const int tid  = threadIdx.x;
    const int lane = tid & 31;
    const int wv   = tid >> 5;
    const int rw   = (wv & 3) << 4;   // wave's 16-row slab in tile
    const int cw   = (wv >> 2) << 5;  // wave's 32-col slab in tile
    const int hf   = lane >> 4;       // lane half (0/1)
    const int l16  = lane & 15;

    // ---- staging assignments ----
    const int m_a  = tid >> 2;              // A: row, 8 consecutive k
    const int kq_a = (tid & 3) << 3;
    const float* aptr = A + (size_t)(MB + m_a) * K + kq_a;

    int n_b = 0, kq_b = 0, kk_b = 0, nq_b = 0;
    bool bvalid = true;
    const float* bptr;
    if (BT) {
        n_b  = tid >> 2;                    // B row = output col, 8 consecutive k
        kq_b = (tid & 3) << 3;
        const int gn = NB + n_b;
        bvalid = (gn < N);
        bptr = B + (size_t)(bvalid ? gn : 0) * K + kq_b;
    } else {
        kk_b = tid >> 3;                    // one k, 8 consecutive n
        nq_b = (tid & 7) << 3;
        bptr = B + (size_t)kk_b * N + NB + nq_b;   // N % 8 == 0 on this path
    }

    // ---- prefetch first K-tile into registers ----
    float4 pa0 = ((const float4*)aptr)[0];
    float4 pa1 = ((const float4*)aptr)[1];
    aptr += 32;
    float4 pb0 = ((const float4*)bptr)[0];
    float4 pb1 = ((const float4*)bptr)[1];
    bptr += BT ? (size_t)32 : (size_t)32 * N;

    v8f acc0 = {};
    v8f acc1 = {};

    for (int KB = 0; KB < K; KB += 32) {
        // ---- write staged registers to LDS (f32 -> f16) ----
        {
            v8h ha;
            ha[0] = (_Float16)pa0.x; ha[1] = (_Float16)pa0.y;
            ha[2] = (_Float16)pa0.z; ha[3] = (_Float16)pa0.w;
            ha[4] = (_Float16)pa1.x; ha[5] = (_Float16)pa1.y;
            ha[6] = (_Float16)pa1.z; ha[7] = (_Float16)pa1.w;
            *(v8h*)&As[m_a][kq_a] = ha;
        }
        if (BT) {
            v8h hb;
            if (bvalid) {
                hb[0] = (_Float16)pb0.x; hb[1] = (_Float16)pb0.y;
                hb[2] = (_Float16)pb0.z; hb[3] = (_Float16)pb0.w;
                hb[4] = (_Float16)pb1.x; hb[5] = (_Float16)pb1.y;
                hb[6] = (_Float16)pb1.z; hb[7] = (_Float16)pb1.w;
            } else {
                #pragma unroll
                for (int j = 0; j < 8; ++j) hb[j] = (_Float16)0.0f;
            }
            *(v8h*)&Bs[n_b][kq_b] = hb;
        } else {
            const float tb[8] = {pb0.x, pb0.y, pb0.z, pb0.w, pb1.x, pb1.y, pb1.z, pb1.w};
            #pragma unroll
            for (int j = 0; j < 8; ++j) Bs[nq_b + j][kk_b] = (_Float16)tb[j];
        }
        __syncthreads();

        // ---- prefetch next K-tile (loads in flight during WMMA) ----
        if (KB + 32 < K) {
            pa0 = ((const float4*)aptr)[0];
            pa1 = ((const float4*)aptr)[1];
            aptr += 32;
            pb0 = ((const float4*)bptr)[0];
            pb1 = ((const float4*)bptr)[1];
            bptr += BT ? (size_t)32 : (size_t)32 * N;
        }

        // ---- A fragment: lane m = rw+l16; K chunks [8*hf,+8) and [16+8*hf,+8)
        const _Float16* ap = &As[rw + l16][0];
        v8h alo = *(const v8h*)(ap + 8 * hf);
        v8h ahi = *(const v8h*)(ap + 16 + 8 * hf);
        v16h af;
        #pragma unroll
        for (int e = 0; e < 8; ++e) { af[e] = alo[e]; af[8 + e] = ahi[e]; }

        // ---- B fragments: lane n = cw(+16)+l16; K chunks [8*hf,+8) and [16+8*hf,+8)
        const _Float16* bp0 = &Bs[cw + l16][0];
        v8h b0lo = *(const v8h*)(bp0 + 8 * hf);
        v8h b0hi = *(const v8h*)(bp0 + 16 + 8 * hf);
        const _Float16* bp1 = &Bs[cw + 16 + l16][0];
        v8h b1lo = *(const v8h*)(bp1 + 8 * hf);
        v8h b1hi = *(const v8h*)(bp1 + 16 + 8 * hf);
        v16h bf0, bf1;
        #pragma unroll
        for (int e = 0; e < 8; ++e) {
            bf0[e] = b0lo[e]; bf0[8 + e] = b0hi[e];
            bf1[e] = b1lo[e]; bf1[8 + e] = b1hi[e];
        }

        acc0 = __builtin_amdgcn_wmma_f32_16x16x32_f16(false, af, false, bf0,
                                                      (short)0, acc0, false, false);
        acc1 = __builtin_amdgcn_wmma_f32_16x16x32_f16(false, af, false, bf1,
                                                      (short)0, acc1, false, false);
        __syncthreads();
    }

    // ---- epilogue: D layout lane n=l16, VGPR v -> row v + 8*hf ----
    const int col0  = NB + cw + l16;
    const int col1  = col0 + 16;
    const int rbase = MB + rw + (hf << 3);
    const float b0 = (bias && col0 < N) ? bias[col0] : 0.0f;
    const float b1 = (bias && col1 < N) ? bias[col1] : 0.0f;
    #pragma unroll
    for (int vv = 0; vv < 8; ++vv) {
        int row = rbase + vv;
        float x0 = acc0[vv] + b0;
        float x1 = acc1[vv] + b1;
        if (ACT == 1) { x0 = gelu_exact(x0); x1 = gelu_exact(x1); }
        if (col0 < N) C[(size_t)row * N + col0] = x0;
        if (col1 < N) C[(size_t)row * N + col1] = x1;
    }
}

// ---------------------------------------------------------------------------
// Row reductions / LayerNorm helpers (256-thread blocks, 8 waves)
// ---------------------------------------------------------------------------
__device__ __forceinline__ float block_sum(float v, float* tmp) {
    #pragma unroll
    for (int o = 16; o; o >>= 1) v += __shfl_xor(v, o, 32);
    if ((threadIdx.x & 31) == 0) tmp[threadIdx.x >> 5] = v;
    __syncthreads();
    float r = (threadIdx.x < 8) ? tmp[threadIdx.x] : 0.0f;
    if (threadIdx.x < 32) {
        #pragma unroll
        for (int o = 16; o; o >>= 1) r += __shfl_xor(r, o, 32);
        if (threadIdx.x == 0) tmp[0] = r;
    }
    __syncthreads();
    float out = tmp[0];
    __syncthreads();
    return out;
}

// out = LN(base + delta) * g + b   (base may be nullptr -> LN(delta))
__global__ __launch_bounds__(256) void add_ln_kernel(const float* __restrict__ base,
                                                     const float* __restrict__ delta,
                                                     const float* __restrict__ g,
                                                     const float* __restrict__ b,
                                                     float* __restrict__ out, int D) {
    __shared__ float tmp[8];
    const size_t ro = (size_t)blockIdx.x * D;
    float s = 0.0f;
    for (int i = threadIdx.x; i < D; i += 256) {
        float t = delta[ro + i] + (base ? base[ro + i] : 0.0f);
        out[ro + i] = t;
        s += t;
    }
    const float mean = block_sum(s, tmp) / (float)D;
    float vs = 0.0f;
    for (int i = threadIdx.x; i < D; i += 256) {
        float d = out[ro + i] - mean;
        vs += d * d;
    }
    const float var = block_sum(vs, tmp) / (float)D;
    const float inv = rsqrtf(var + 1e-5f);
    for (int i = threadIdx.x; i < D; i += 256)
        out[ro + i] = (out[ro + i] - mean) * inv * g[i] + b[i];
}

__global__ __launch_bounds__(256) void embed_pos_kernel(const int* __restrict__ ids,
                                                        const float* __restrict__ emb,
                                                        const float* __restrict__ pos,
                                                        float* __restrict__ out, int D) {
    const int row = blockIdx.x;
    const size_t e = (size_t)ids[row] * D;
    const size_t ro = (size_t)row * D;
    for (int i = threadIdx.x; i < D; i += 256)
        out[ro + i] = emb[e + i] + pos[ro + i];
}

// out[N] = x0[K] @ W[K,N] + b  (tiny GEMV for qg: single row)
__global__ __launch_bounds__(256) void vecmat_kernel(const float* __restrict__ x0,
                                                     const float* __restrict__ W,
                                                     const float* __restrict__ b,
                                                     float* __restrict__ out,
                                                     int K, int N) {
    const int o = blockIdx.x * 256 + threadIdx.x;
    if (o >= N) return;
    float acc = b[o];
    for (int k = 0; k < K; ++k) acc += x0[k] * W[(size_t)k * N + o];
    out[o] = acc;
}

// ---------------------------------------------------------------------------
// Longformer sliding-window attention (W1=64 -> 129 window slots + global k[0])
// One wave per (head, query). Writes merged [S, H*HD].
// ---------------------------------------------------------------------------
__global__ __launch_bounds__(32) void enc_win_attn_kernel(const float* __restrict__ q,
                                                          const float* __restrict__ k,
                                                          const float* __restrict__ v,
                                                          float* __restrict__ out, int S) {
    const int h = blockIdx.x, s = blockIdx.y, lane = threadIdx.x;
    __shared__ float sc[132];
    const size_t qo = (size_t)s * D_MODEL + h * HDIM;
    const float q0 = q[qo + lane] * ATT_SCALE;
    const float q1 = q[qo + 32 + lane] * ATT_SCALE;

    for (int j = 0; j < 130; ++j) {
        int pos; bool valid;
        if (j == 0) { pos = 0; valid = true; }                // global slot: k[0]
        else { pos = s + j - 65; valid = (pos >= 0) && (pos < S) && (pos != 0); }
        int pc = pos < 0 ? 0 : (pos >= S ? S - 1 : pos);
        const size_t ko = (size_t)pc * D_MODEL + h * HDIM;
        float part = q0 * k[ko + lane] + q1 * k[ko + 32 + lane];
        #pragma unroll
        for (int o = 16; o; o >>= 1) part += __shfl_xor(part, o, 32);
        if (lane == 0) sc[j] = valid ? part : NEGBIG;
    }
    __syncthreads();
    float mx = -1e30f;
    for (int j = lane; j < 130; j += 32) mx = fmaxf(mx, sc[j]);
    #pragma unroll
    for (int o = 16; o; o >>= 1) mx = fmaxf(mx, __shfl_xor(mx, o, 32));
    float sum = 0.0f;
    for (int j = lane; j < 130; j += 32) { float e = expf(sc[j] - mx); sc[j] = e; sum += e; }
    #pragma unroll
    for (int o = 16; o; o >>= 1) sum += __shfl_xor(sum, o, 32);
    __syncthreads();
    const float inv = 1.0f / sum;
    float a0 = 0.0f, a1 = 0.0f;
    for (int j = 0; j < 130; ++j) {
        int pos = (j == 0) ? 0 : s + j - 65;
        int pc = pos < 0 ? 0 : (pos >= S ? S - 1 : pos);
        const size_t vo = (size_t)pc * D_MODEL + h * HDIM;
        const float p = sc[j] * inv;   // masked slots are exactly 0
        a0 += p * v[vo + lane];
        a1 += p * v[vo + 32 + lane];
    }
    out[qo + lane] = a0;
    out[qo + 32 + lane] = a1;
}

// Global-token attention (row 0 of encoder output). One wave per head.
__global__ __launch_bounds__(32) void enc_global_attn_kernel(const float* __restrict__ qg,
                                                             const float* __restrict__ kg,
                                                             const float* __restrict__ vg,
                                                             float* __restrict__ out, int S) {
    const int h = blockIdx.x, lane = threadIdx.x;
    __shared__ float sc[S_ENC];
    __shared__ float qr[HDIM];
    qr[lane]      = qg[h * HDIM + lane] * ATT_SCALE;
    qr[lane + 32] = qg[h * HDIM + 32 + lane] * ATT_SCALE;
    __syncthreads();
    for (int s = lane; s < S; s += 32) {
        const float* kp = kg + (size_t)s * D_MODEL + h * HDIM;
        float d = 0.0f;
        #pragma unroll 8
        for (int i = 0; i < HDIM; ++i) d += qr[i] * kp[i];
        sc[s] = d;
    }
    __syncthreads();
    float mx = -1e30f;
    for (int s = lane; s < S; s += 32) mx = fmaxf(mx, sc[s]);
    #pragma unroll
    for (int o = 16; o; o >>= 1) mx = fmaxf(mx, __shfl_xor(mx, o, 32));
    float sum = 0.0f;
    for (int s = lane; s < S; s += 32) { float e = expf(sc[s] - mx); sc[s] = e; sum += e; }
    #pragma unroll
    for (int o = 16; o; o >>= 1) sum += __shfl_xor(sum, o, 32);
    __syncthreads();
    const float inv = 1.0f / sum;
    float a0 = 0.0f, a1 = 0.0f;
    for (int s = 0; s < S; ++s) {
        const float* vp = vg + (size_t)s * D_MODEL + h * HDIM;
        const float p = sc[s] * inv;
        a0 += p * vp[lane];
        a1 += p * vp[lane + 32];
    }
    out[h * HDIM + lane] = a0;          // row 0 of merged output
    out[h * HDIM + 32 + lane] = a1;
}

// Dense attention (decoder self w/ causal mask, or cross). One wave per (h,t).
__global__ __launch_bounds__(32) void dec_attn_kernel(const float* __restrict__ q,
                                                      const float* __restrict__ k,
                                                      const float* __restrict__ v,
                                                      float* __restrict__ out,
                                                      int Skv, int causal) {
    const int h = blockIdx.x, t = blockIdx.y, lane = threadIdx.x;
    __shared__ float sc[S_ENC];
    __shared__ float qr[HDIM];
    const size_t qo = (size_t)t * D_MODEL + h * HDIM;
    qr[lane]      = q[qo + lane] * ATT_SCALE;
    qr[lane + 32] = q[qo + 32 + lane] * ATT_SCALE;
    __syncthreads();
    const int lim = causal ? (t + 1) : Skv;
    for (int s = lane; s < lim; s += 32) {
        const float* kp = k + (size_t)s * D_MODEL + h * HDIM;
        float d = 0.0f;
        #pragma unroll 8
        for (int i = 0; i < HDIM; ++i) d += qr[i] * kp[i];
        sc[s] = d;
    }
    __syncthreads();
    float mx = -1e30f;
    for (int s = lane; s < lim; s += 32) mx = fmaxf(mx, sc[s]);
    #pragma unroll
    for (int o = 16; o; o >>= 1) mx = fmaxf(mx, __shfl_xor(mx, o, 32));
    float sum = 0.0f;
    for (int s = lane; s < lim; s += 32) { float e = expf(sc[s] - mx); sc[s] = e; sum += e; }
    #pragma unroll
    for (int o = 16; o; o >>= 1) sum += __shfl_xor(sum, o, 32);
    __syncthreads();
    const float inv = 1.0f / sum;
    float a0 = 0.0f, a1 = 0.0f;
    for (int s = 0; s < lim; ++s) {
        const float* vp = v + (size_t)s * D_MODEL + h * HDIM;
        const float p = sc[s] * inv;
        a0 += p * vp[lane];
        a1 += p * vp[lane + 32];
    }
    out[qo + lane] = a0;
    out[qo + 32 + lane] = a1;
}

// ---------------------------------------------------------------------------
// Host orchestration
// ---------------------------------------------------------------------------
extern "C" void kernel_launch(void* const* d_in, const int* in_sizes, int n_in,
                              void* d_out, int out_size, void* d_ws, size_t ws_size,
                              hipStream_t stream) {
    (void)in_sizes; (void)n_in; (void)out_size; (void)ws_size;

    const int*   enc_ids  = (const int*)d_in[0];
    const int*   dec_ids  = (const int*)d_in[1];
    int idx = 2;
    const float* embed    = (const float*)d_in[idx++];
    const float* enc_pos  = (const float*)d_in[idx++];
    const float* dec_pos  = (const float*)d_in[idx++];
    const float* enc_ln_g = (const float*)d_in[idx++];
    const float* enc_ln_b = (const float*)d_in[idx++];
    const float* dec_ln_g = (const float*)d_in[idx++];
    const float* dec_ln_b = (const float*)d_in[idx++];
    // enc layer: q_w,q_b,k_w,k_b,v_w,v_b,qg_w,qg_b,kg_w,kg_b,vg_w,vg_b,o_w,o_b,
    //            ln1_g,ln1_b,fc1_w,fc1_b,fc2_w,fc2_b,ln2_g,ln2_b  (22)
    const float* EL[2][22];
    for (int l = 0; l < 2; ++l)
        for (int j = 0; j < 22; ++j) EL[l][j] = (const float*)d_in[idx++];
    // dec layer: sq_w..so_b (8), cq_w..co_b (8), ln1_g,ln1_b, ln2_g,ln2_b,
    //            fc1_w,fc1_b,fc2_w,fc2_b, ln3_g,ln3_b  (26)
    const float* DL[2][26];
    for (int l = 0; l < 2; ++l)
        for (int j = 0; j < 26; ++j) DL[l][j] = (const float*)d_in[idx++];

    // -------- workspace bump allocation --------
    size_t off = 0;
    auto alloc = [&](size_t elems) -> float* {
        float* p = (float*)((char*)d_ws + off);
        off += (elems * sizeof(float) + 255) & ~(size_t)255;
        return p;
    };
    float* xA   = alloc((size_t)S_ENC * D_MODEL);
    float* xB   = alloc((size_t)S_ENC * D_MODEL);
    float* qb   = alloc((size_t)S_ENC * D_MODEL);
    float* kb   = alloc((size_t)S_ENC * D_MODEL);
    float* vb   = alloc((size_t)S_ENC * D_MODEL);
    float* kgb  = alloc((size_t)S_ENC * D_MODEL);
    float* vgb  = alloc((size_t)S_ENC * D_MODEL);
    float* attn = alloc((size_t)S_ENC * D_MODEL);
    float* proj = alloc((size_t)S_ENC * D_MODEL);
    float* hbuf = alloc((size_t)S_ENC * FFN);
    float* qg   = alloc(D_MODEL);
    float* yA   = alloc((size_t)T_DEC * D_MODEL);
    float* yB   = alloc((size_t)T_DEC * D_MODEL);

    auto gemm = [&](const float* A, const float* W, const float* bias, float* Cp,
                    int M, int N, int K, int act) {
        dim3 g(M / 64, (N + 63) / 64);     // x = row-block (B-stationary via L2)
        if (act) gemm_kernel<1, false><<<g, 256, 0, stream>>>(A, W, bias, Cp, M, N, K);
        else     gemm_kernel<0, false><<<g, 256, 0, stream>>>(A, W, bias, Cp, M, N, K);
    };

    // ================= Encoder =================
    embed_pos_kernel<<<S_ENC, 256, 0, stream>>>(enc_ids, embed, enc_pos, proj, D_MODEL);
    add_ln_kernel<<<S_ENC, 256, 0, stream>>>(nullptr, proj, enc_ln_g, enc_ln_b, xA, D_MODEL);
    float* cur = xA; float* nxt = xB;

    for (int l = 0; l < 2; ++l) {
        const float* const* L = EL[l];
        gemm(cur, L[0], L[1], qb, S_ENC, D_MODEL, D_MODEL, 0);   // q
        gemm(cur, L[2], L[3], kb, S_ENC, D_MODEL, D_MODEL, 0);   // k
        gemm(cur, L[4], L[5], vb, S_ENC, D_MODEL, D_MODEL, 0);   // v
        enc_win_attn_kernel<<<dim3(NHEAD, S_ENC), 32, 0, stream>>>(qb, kb, vb, attn, S_ENC);
        vecmat_kernel<<<D_MODEL / 256, 256, 0, stream>>>(cur, L[6], L[7], qg, D_MODEL, D_MODEL); // qg from x[0]
        gemm(cur, L[8],  L[9],  kgb, S_ENC, D_MODEL, D_MODEL, 0);  // kg
        gemm(cur, L[10], L[11], vgb, S_ENC, D_MODEL, D_MODEL, 0);  // vg
        enc_global_attn_kernel<<<NHEAD, 32, 0, stream>>>(qg, kgb, vgb, attn, S_ENC); // overwrite row 0
        gemm(attn, L[12], L[13], proj, S_ENC, D_MODEL, D_MODEL, 0); // o proj
        add_ln_kernel<<<S_ENC, 256, 0, stream>>>(cur, proj, L[14], L[15], nxt, D_MODEL);
        { float* t = cur; cur = nxt; nxt = t; }
        gemm(cur, L[16], L[17], hbuf, S_ENC, FFN, D_MODEL, 1);      // fc1 + GELU
        gemm(hbuf, L[18], L[19], proj, S_ENC, D_MODEL, FFN, 0);     // fc2
        add_ln_kernel<<<S_ENC, 256, 0, stream>>>(cur, proj, L[20], L[21], nxt, D_MODEL);
        { float* t = cur; cur = nxt; nxt = t; }
    }
    const float* encx = cur;   // encoder output [S_ENC, D]

    // ================= Decoder =================
    embed_pos_kernel<<<T_DEC, 256, 0, stream>>>(dec_ids, embed, dec_pos, proj, D_MODEL);
    add_ln_kernel<<<T_DEC, 256, 0, stream>>>(nullptr, proj, dec_ln_g, dec_ln_b, yA, D_MODEL);
    float* ycur = yA; float* ynxt = yB;

    for (int l = 0; l < 2; ++l) {
        const float* const* L = DL[l];
        // ---- causal self-attention ----
        gemm(ycur, L[0], L[1], qb, T_DEC, D_MODEL, D_MODEL, 0);
        gemm(ycur, L[2], L[3], kb, T_DEC, D_MODEL, D_MODEL, 0);
        gemm(ycur, L[4], L[5], vb, T_DEC, D_MODEL, D_MODEL, 0);
        dec_attn_kernel<<<dim3(NHEAD, T_DEC), 32, 0, stream>>>(qb, kb, vb, attn, T_DEC, 1);
        gemm(attn, L[6], L[7], proj, T_DEC, D_MODEL, D_MODEL, 0);
        add_ln_kernel<<<T_DEC, 256, 0, stream>>>(ycur, proj, L[16], L[17], ynxt, D_MODEL);
        { float* t = ycur; ycur = ynxt; ynxt = t; }
        // ---- cross-attention ----
        gemm(ycur, L[8],  L[9],  qb,  T_DEC, D_MODEL, D_MODEL, 0);
        gemm(encx, L[10], L[11], kgb, S_ENC, D_MODEL, D_MODEL, 0);
        gemm(encx, L[12], L[13], vgb, S_ENC, D_MODEL, D_MODEL, 0);
        dec_attn_kernel<<<dim3(NHEAD, T_DEC), 32, 0, stream>>>(qb, kgb, vgb, attn, S_ENC, 0);
        gemm(attn, L[14], L[15], proj, T_DEC, D_MODEL, D_MODEL, 0);
        add_ln_kernel<<<T_DEC, 256, 0, stream>>>(ycur, proj, L[18], L[19], ynxt, D_MODEL);
        { float* t = ycur; ycur = ynxt; ynxt = t; }
        // ---- FFN ----
        gemm(ycur, L[20], L[21], hbuf, T_DEC, FFN, D_MODEL, 1);
        gemm(hbuf, L[22], L[23], proj, T_DEC, D_MODEL, FFN, 0);
        add_ln_kernel<<<T_DEC, 256, 0, stream>>>(ycur, proj, L[24], L[25], ynxt, D_MODEL);
        { float* t = ycur; ycur = ynxt; ynxt = t; }
    }

    // ================= Logits: y @ embed.T =================
    {
        dim3 g(T_DEC / 64, (VOCAB + 63) / 64);
        gemm_kernel<0, true><<<g, 256, 0, stream>>>(ycur, embed, nullptr,
                                                    (float*)d_out, T_DEC, VOCAB, D_MODEL);
    }
}
